// VanillaConv_25323127177742
// MI455X (gfx1250) — matrix operand order repaired and checked by reading, not compiled, not run on hardware
//
#include <hip/hip_runtime.h>

#define NTOK 2304      // 48*48
#define CCH  256
#define BATCH 2
#define HEADS 8
#define DHEAD 64
#define HIDDEN 512
#define SCALE_Q 0.125f

typedef __attribute__((ext_vector_type(16))) __bf16        v16bf;
typedef __attribute__((ext_vector_type(8)))  float         v8f;
typedef __attribute__((ext_vector_type(8)))  unsigned int  v8u;
typedef __attribute__((ext_vector_type(4)))  unsigned int  u32x4;
typedef __attribute__((ext_vector_type(8)))  int           i32x8;
typedef __attribute__((ext_vector_type(4)))  int           i32x4;

// ---------- helpers ----------

__device__ __forceinline__ unsigned short f2bf(float f) {
  unsigned int u = __builtin_bit_cast(unsigned int, f);
  u += 0x7FFFu + ((u >> 16) & 1u);              // round-to-nearest-even
  return (unsigned short)(u >> 16);
}

__device__ __forceinline__ v8f wmma_bf16(v16bf a, v16bf b, v8f c) {
  // v_wmma_f32_16x16x32_bf16  D = A(16x32) * B(32x16) + C
  return __builtin_amdgcn_wmma_f32_16x16x32_bf16(
      false, a, false, b, (short)0, c, false, false);
}

// A fragment, 16x32 bf16, from row-major src (pre-offset to (row0,col0)), row
// stride ld elements.  M = lane&15 for both halves (ISA); this code's K order: lane half
// selects K subrange {0..15} vs {16..31}; one contiguous 32B run.
__device__ __forceinline__ v16bf load_a_frag(const unsigned short* src, int ld, int lane) {
  const int row = lane & 15;
  const int kb  = (lane >> 4) * 16;
  v8u t;
#pragma unroll
  for (int v = 0; v < 8; ++v) {
    int k = kb + 2 * v;
    t[v] = *(const unsigned int*)(src + (size_t)row * ld + k);
  }
  return __builtin_bit_cast(v16bf, t);
}

// B fragment, 32x16 bf16, element (k,n) = src[n*ld + k]  (K contiguous per
// column -> one 32-byte per-lane vector load; lowers to 2x ds/global b128).
// N = lane&15; this code's K order (same as A): lanes 0-15 hold K=0..15, lanes 16-31 hold K=16..31.
__device__ __forceinline__ v16bf load_b_frag_cols(const unsigned short* src, int ld, int lane) {
  const int col = lane & 15;
  const int kb  = (lane >> 4) * 16;
  v8u t = *(const v8u*)(src + (size_t)col * ld + kb);
  return __builtin_bit_cast(v16bf, t);
}

__device__ __forceinline__ unsigned int lds_off(const void* p) {
  // flat LDS addresses carry the LDS byte offset in the low 32 bits
  return (unsigned int)(unsigned long long)(uintptr_t)p;
}

// Tensor Data Mover: 2D tile load (tile_dim1 rows of tile_dim0 contiguous
// elements, row stride stride0) of bf16 data into LDS. D# per ISA 08 §8.
__device__ __forceinline__ void tdm_load_2d(unsigned int ldsaddr,
                                            const unsigned short* gsrc,
                                            unsigned int tensor_dim0,
                                            unsigned int tensor_dim1,
                                            unsigned int tile_dim0,
                                            unsigned int tile_dim1,
                                            unsigned int stride0) {
  unsigned long long ga = (unsigned long long)(uintptr_t)gsrc;
  u32x4 g0;
  g0[0] = 1u;                                   // count=1, user descriptor
  g0[1] = ldsaddr;                              // lds_addr
  g0[2] = (unsigned int)ga;                     // global_addr[31:0]
  g0[3] = (unsigned int)((ga >> 32) & 0x1FFFFFFu) | (2u << 30);  // addr[56:32] | type=2
  i32x8 g1;
  g1[0] = (int)(1u << 16);                                          // data_size=2B
  g1[1] = (int)((tensor_dim0 & 0xFFFFu) << 16);                     // tensor_dim0 lo
  g1[2] = (int)(((tensor_dim0 >> 16) & 0xFFFFu) | ((tensor_dim1 & 0xFFFFu) << 16));
  g1[3] = (int)(((tensor_dim1 >> 16) & 0xFFFFu) | ((tile_dim0 & 0xFFFFu) << 16));
  g1[4] = (int)(tile_dim1 & 0xFFFFu);                               // tile_dim2=0
  g1[5] = (int)stride0;                                             // dim0_stride lo32
  g1[6] = 0;
  g1[7] = 0;
  i32x4 z4 = {0, 0, 0, 0};
  i32x8 z8 = {0, 0, 0, 0, 0, 0, 0, 0};
  __builtin_amdgcn_tensor_load_to_lds(g0, g1, z4, z4, z8, 0);
}

// ---------- kernel 1: per-batch mean / rsigma ----------

__global__ void stats_kernel(const float* __restrict__ x, float* __restrict__ stats) {
  const int b = blockIdx.x;
  const int tid = threadIdx.x;
  const float* xb = x + (size_t)b * CCH * NTOK;
  float s = 0.f, ss = 0.f;
  for (size_t i = tid; i < (size_t)CCH * NTOK; i += blockDim.x) {
    float v = xb[i];
    s += v; ss += v * v;
  }
  __shared__ float sh[256], sh2[256];
  sh[tid] = s; sh2[tid] = ss;
  __syncthreads();
  for (int st = 128; st > 0; st >>= 1) {
    if (tid < st) { sh[tid] += sh[tid + st]; sh2[tid] += sh2[tid + st]; }
    __syncthreads();
  }
  if (tid == 0) {
    float inv_n = 1.f / ((float)CCH * (float)NTOK);
    float mu = sh[0] * inv_n;
    float var = sh2[0] * inv_n - mu * mu;
    stats[b * 2 + 0] = mu;
    stats[b * 2 + 1] = rsqrtf(var + 1e-5f);
  }
}

// ---------- kernel 2: f32 -> bf16 convert (weights) ----------

__global__ void convert_bf16_kernel(const float* __restrict__ src,
                                    unsigned short* __restrict__ dst, int n) {
  int i = blockIdx.x * blockDim.x + threadIdx.x;
  if (i < n) dst[i] = f2bf(src[i]);
}

// ---------- kernel 3: normalize + affine -> bf16, transposed (B, N, C) ----------

__global__ void norm_kernel(const float* __restrict__ x,
                            const float* __restrict__ gw,
                            const float* __restrict__ gb,
                            const float* __restrict__ stats,
                            unsigned short* __restrict__ xnT) {
  size_t i = (size_t)blockIdx.x * blockDim.x + threadIdx.x;
  if (i >= (size_t)BATCH * CCH * NTOK) return;
  int b = (int)(i / ((size_t)CCH * NTOK));
  int c = (int)((i / NTOK) % CCH);
  int n = (int)(i % NTOK);
  float v = (x[i] - stats[b * 2]) * stats[b * 2 + 1] * gw[c] + gb[c];
  xnT[((size_t)b * NTOK + n) * CCH + c] = f2bf(v);
}

// ---------- kernel 4: QKV GEMM (TDM-staged WMMA bf16) ----------
// qkv[o,n] = sum_c w_in[o,c] * xnT[n,c];  X tiles (64N x 128K, K contiguous)
// streamed to LDS by the TDM; scatter into Q,K (n-major) / V (d-major)

__global__ __launch_bounds__(128) void qkv_gemm_kernel(
    const unsigned short* __restrict__ Wb,   // 1536 x 256 bf16 row-major
    const unsigned short* __restrict__ XnT,  // B x NTOK x 256 bf16
    unsigned short* __restrict__ Qw,         // B x H x NTOK x 64
    unsigned short* __restrict__ Kw,         // B x H x NTOK x 64
    unsigned short* __restrict__ Vw) {       // B x H x 64 x NTOK
  __shared__ __attribute__((aligned(32))) unsigned short lds_x[64 * 128];  // [n][k]
  const int tid  = threadIdx.x;
  const int wave = tid >> 5;
  const int lane = tid & 31;
  const int hl   = lane >> 4;
  const int l16  = lane & 15;
  const int n0 = blockIdx.x * 64;          // block n base (4 waves x 16)
  const int o0 = blockIdx.y * 16;
  const int b  = blockIdx.z;
  const unsigned short* xbT = XnT + (size_t)b * NTOK * CCH;

  v8f acc = {0.f, 0.f, 0.f, 0.f, 0.f, 0.f, 0.f, 0.f};
#pragma unroll
  for (int kc = 0; kc < CCH; kc += 128) {
    if (tid == 0) {
      tdm_load_2d(lds_off(lds_x), xbT + (size_t)n0 * CCH + kc,
                  CCH, NTOK, 128, 64, CCH);
      __builtin_amdgcn_s_wait_tensorcnt(0);
    }
    __syncthreads();
#pragma unroll
    for (int k = 0; k < 128; k += 32) {
      v16bf a  = load_a_frag(Wb + (size_t)o0 * CCH + kc + k, CCH, lane);
      v16bf bb = load_b_frag_cols(lds_x + (size_t)(wave * 16) * 128 + k, 128, lane);
      acc = wmma_bf16(a, bb, acc);
    }
    __syncthreads();   // tile consumed before next TDM overwrite
  }
  const int sec = o0 >> 9;   // 0=q 1=k 2=v (whole 16-row tile in one section)
#pragma unroll
  for (int r = 0; r < 8; ++r) {
    int o  = o0 + r + 8 * hl;
    int c2 = o & 511;
    int hh = c2 >> 6;
    int d  = c2 & 63;
    int n  = n0 + wave * 16 + l16;
    float val = acc[r];
    if (sec == 0)
      Qw[(((size_t)(b * HEADS + hh)) * NTOK + n) * DHEAD + d] = f2bf(val * SCALE_Q);
    else if (sec == 1)
      Kw[(((size_t)(b * HEADS + hh)) * NTOK + n) * DHEAD + d] = f2bf(val);
    else
      Vw[((size_t)(b * HEADS + hh)) * DHEAD * NTOK + (size_t)d * NTOK + n] = f2bf(val);
  }
}

// ---------- kernel 5: flash attention (TDM + WMMA) ----------
// One block = 4 waves = 64 queries of one (b,h); streams 32-key tiles of K,V
// staged into LDS by the Tensor Data Mover.

__global__ __launch_bounds__(128) void attn_kernel(
    const unsigned short* __restrict__ Q,    // B x H x NTOK x 64
    const unsigned short* __restrict__ K,    // B x H x NTOK x 64
    const unsigned short* __restrict__ V,    // B x H x 64 x NTOK
    unsigned short* __restrict__ O2T) {      // B x NTOK x 512
  __shared__ __attribute__((aligned(32))) unsigned short lds_k[32 * 64];     // [j][d]
  __shared__ __attribute__((aligned(32))) unsigned short lds_v[64 * 32];     // [d][j]
  __shared__ __attribute__((aligned(32))) unsigned short lds_p[4][16 * 32];  // [i][j]

  const int tid  = threadIdx.x;
  const int wave = tid >> 5;
  const int lane = tid & 31;
  const int hl   = lane >> 4;
  const int l16  = lane & 15;

  const int h = blockIdx.y;
  const int b = blockIdx.z;
  const int i0 = blockIdx.x * 64 + wave * 16;

  __builtin_amdgcn_s_cluster_barrier();   // NOP when not dispatched as cluster

  const unsigned short* qbase = Q + (((size_t)(b * HEADS + h)) * NTOK + i0) * DHEAD;
  const unsigned short* kbase = K + ((size_t)(b * HEADS + h)) * NTOK * DHEAD;
  const unsigned short* vbase = V + ((size_t)(b * HEADS + h)) * DHEAD * NTOK;

  // Q A-fragments (16 queries x 64 d), resident for whole kernel
  v16bf aq0 = load_a_frag(qbase, DHEAD, lane);        // d 0..31
  v16bf aq1 = load_a_frag(qbase + 32, DHEAD, lane);   // d 32..63

  v8f oacc[4];
#pragma unroll
  for (int dd = 0; dd < 4; ++dd)
    oacc[dd] = (v8f){0.f, 0.f, 0.f, 0.f, 0.f, 0.f, 0.f, 0.f};
  float m[8], l[8];
#pragma unroll
  for (int r = 0; r < 8; ++r) { m[r] = -1e30f; l[r] = 0.f; }

  for (int jt = 0; jt < NTOK / 32; ++jt) {
    const int j0 = jt * 32;
    if (tid == 0) {
      // K: 32 key-rows x 64 contiguous d ; V: 64 d-rows x 32 keys
      tdm_load_2d(lds_off(lds_k), kbase + (size_t)j0 * DHEAD,
                  DHEAD, NTOK, DHEAD, 32, DHEAD);
      tdm_load_2d(lds_off(lds_v), vbase + j0, NTOK, DHEAD, 32, DHEAD, NTOK);
      __builtin_amdgcn_s_wait_tensorcnt(0);
    }
    __syncthreads();

    // S(16x32) = Qtile(16x64) . Ktile^T ; B elem (K=d, N=j) = lds_k[j*64 + d]
    v8f s0 = {0.f, 0.f, 0.f, 0.f, 0.f, 0.f, 0.f, 0.f};
    v8f s1 = {0.f, 0.f, 0.f, 0.f, 0.f, 0.f, 0.f, 0.f};
    {
      v16bf bk00 = load_b_frag_cols(lds_k + 0,             64, lane);  // j 0-15,  d 0-31
      v16bf bk10 = load_b_frag_cols(lds_k + 32,            64, lane);  // j 0-15,  d 32-63
      v16bf bk01 = load_b_frag_cols(lds_k + 16 * 64,       64, lane);  // j 16-31, d 0-31
      v16bf bk11 = load_b_frag_cols(lds_k + 16 * 64 + 32,  64, lane);  // j 16-31, d 32-63
      s0 = wmma_bf16(aq0, bk00, s0);
      s0 = wmma_bf16(aq1, bk10, s0);
      s1 = wmma_bf16(aq0, bk01, s1);
      s1 = wmma_bf16(aq1, bk11, s1);
    }

    // streaming softmax update; row r+8*hl lives in lanes of this 16-half
    float sc[8];
#pragma unroll
    for (int r = 0; r < 8; ++r) {
      float rv = fmaxf(s0[r], s1[r]);
      rv = fmaxf(rv, __shfl_xor(rv, 1, 16));
      rv = fmaxf(rv, __shfl_xor(rv, 2, 16));
      rv = fmaxf(rv, __shfl_xor(rv, 4, 16));
      rv = fmaxf(rv, __shfl_xor(rv, 8, 16));
      float mn = fmaxf(m[r], rv);
      sc[r] = __expf(m[r] - mn);
      m[r] = mn;
    }
    v8f p0, p1;
#pragma unroll
    for (int r = 0; r < 8; ++r) {
      p0[r] = __expf(s0[r] - m[r]);
      p1[r] = __expf(s1[r] - m[r]);
      float rs = p0[r] + p1[r];
      rs += __shfl_xor(rs, 1, 16);
      rs += __shfl_xor(rs, 2, 16);
      rs += __shfl_xor(rs, 4, 16);
      rs += __shfl_xor(rs, 8, 16);
      l[r] = l[r] * sc[r] + rs;
    }
#pragma unroll
    for (int dd = 0; dd < 4; ++dd)
#pragma unroll
      for (int r = 0; r < 8; ++r) oacc[dd][r] *= sc[r];

    // D-layout P -> LDS -> A-layout fragment
#pragma unroll
    for (int r = 0; r < 8; ++r) {
      int row = r + 8 * hl;
      lds_p[wave][row * 32 + l16]      = f2bf(p0[r]);
      lds_p[wave][row * 32 + 16 + l16] = f2bf(p1[r]);
    }
    __syncthreads();

    v16bf ap = load_a_frag(&lds_p[wave][0], 32, lane);
    // O(16x64) += P(16x32) . Vtile(32keys x 64d); B elem (K=j, N=d) = lds_v[d*32 + j]
#pragma unroll
    for (int dd = 0; dd < 4; ++dd) {
      v16bf bv = load_b_frag_cols(lds_v + (size_t)(16 * dd) * 32, 32, lane);
      oacc[dd] = wmma_bf16(ap, bv, oacc[dd]);
    }
    __syncthreads();   // lds_k / lds_v fully consumed before next TDM store
  }

  // normalize and scatter to (b, n, h*64+d) bf16 (lane-contiguous in d)
#pragma unroll
  for (int r = 0; r < 8; ++r) {
    float inv = (l[r] > 0.f) ? (1.f / l[r]) : 0.f;
    int i = i0 + r + 8 * hl;
#pragma unroll
    for (int dd = 0; dd < 4; ++dd) {
      int c2 = h * DHEAD + 16 * dd + l16;
      O2T[((size_t)b * NTOK + i) * HIDDEN + c2] = f2bf(oacc[dd][r] * inv);
    }
  }
}

// ---------- kernel 6: out projection GEMM (TDM-staged) + bias + residual ----------

__global__ __launch_bounds__(128) void out_gemm_kernel(
    const unsigned short* __restrict__ Wob,  // 256 x 512 bf16 row-major
    const unsigned short* __restrict__ O2T,  // B x NTOK x 512 bf16
    const float* __restrict__ b_out,
    const float* __restrict__ x,
    float* __restrict__ y) {
  __shared__ __attribute__((aligned(32))) unsigned short lds_x[64 * 128];  // [n][k]
  const int tid  = threadIdx.x;
  const int wave = tid >> 5;
  const int lane = tid & 31;
  const int hl   = lane >> 4;
  const int l16  = lane & 15;
  const int n0 = blockIdx.x * 64;
  const int o0 = blockIdx.y * 16;
  const int b  = blockIdx.z;
  const unsigned short* obT = O2T + (size_t)b * NTOK * HIDDEN;

  v8f acc = {0.f, 0.f, 0.f, 0.f, 0.f, 0.f, 0.f, 0.f};
#pragma unroll
  for (int kc = 0; kc < HIDDEN; kc += 128) {
    if (tid == 0) {
      tdm_load_2d(lds_off(lds_x), obT + (size_t)n0 * HIDDEN + kc,
                  HIDDEN, NTOK, 128, 64, HIDDEN);
      __builtin_amdgcn_s_wait_tensorcnt(0);
    }
    __syncthreads();
#pragma unroll
    for (int k = 0; k < 128; k += 32) {
      v16bf a  = load_a_frag(Wob + (size_t)o0 * HIDDEN + kc + k, HIDDEN, lane);
      v16bf bb = load_b_frag_cols(lds_x + (size_t)(wave * 16) * 128 + k, 128, lane);
      acc = wmma_bf16(a, bb, acc);
    }
    __syncthreads();
  }
#pragma unroll
  for (int r = 0; r < 8; ++r) {
    int o = o0 + r + 8 * hl;
    int n = n0 + wave * 16 + l16;
    size_t idx = ((size_t)b * CCH + o) * NTOK + n;
    y[idx] = acc[r] + b_out[o] + x[idx];
  }
}

// ---------- launcher ----------

extern "C" void kernel_launch(void* const* d_in, const int* in_sizes, int n_in,
                              void* d_out, int out_size, void* d_ws, size_t ws_size,
                              hipStream_t stream) {
  const float* x     = (const float*)d_in[0];
  const float* gn_w  = (const float*)d_in[1];
  const float* gn_b  = (const float*)d_in[2];
  const float* w_in  = (const float*)d_in[3];   // 1536 x 256
  const float* w_out = (const float*)d_in[4];   // 256 x 512
  const float* b_out = (const float*)d_in[5];

  char* ws = (char*)d_ws;
  const size_t OFF_STATS = 0;
  const size_t OFF_WIN   = 256;
  const size_t OFF_WOUT  = OFF_WIN  + (size_t)1536 * 256 * 2;               //  786688
  const size_t OFF_XN    = OFF_WOUT + (size_t)256 * 512 * 2;                // 1048832
  const size_t OFF_Q     = OFF_XN   + (size_t)BATCH * CCH * NTOK * 2;       // 3408128
  const size_t OFF_K     = OFF_Q    + (size_t)BATCH * HIDDEN * NTOK * 2;    // 8126720
  const size_t OFF_V     = OFF_K    + (size_t)BATCH * HIDDEN * NTOK * 2;
  const size_t OFF_O2    = OFF_V    + (size_t)BATCH * HIDDEN * NTOK * 2;

  float*          stats = (float*)(ws + OFF_STATS);
  unsigned short* win_b = (unsigned short*)(ws + OFF_WIN);
  unsigned short* wout_b= (unsigned short*)(ws + OFF_WOUT);
  unsigned short* xnT   = (unsigned short*)(ws + OFF_XN);
  unsigned short* Qw    = (unsigned short*)(ws + OFF_Q);
  unsigned short* Kw    = (unsigned short*)(ws + OFF_K);
  unsigned short* Vw    = (unsigned short*)(ws + OFF_V);
  unsigned short* O2Tw  = (unsigned short*)(ws + OFF_O2);

  stats_kernel<<<BATCH, 256, 0, stream>>>(x, stats);

  {
    int n1 = 1536 * 256;
    convert_bf16_kernel<<<(n1 + 255) / 256, 256, 0, stream>>>(w_in, win_b, n1);
    int n2 = 256 * 512;
    convert_bf16_kernel<<<(n2 + 255) / 256, 256, 0, stream>>>(w_out, wout_b, n2);
  }

  {
    size_t n = (size_t)BATCH * CCH * NTOK;
    norm_kernel<<<(unsigned)((n + 255) / 256), 256, 0, stream>>>(x, gn_w, gn_b, stats, xnT);
  }

  qkv_gemm_kernel<<<dim3(NTOK / 64, 1536 / 16, BATCH), 128, 0, stream>>>(
      win_b, xnT, Qw, Kw, Vw);

  attn_kernel<<<dim3(NTOK / 64, HEADS, BATCH), 128, 0, stream>>>(Qw, Kw, Vw, O2Tw);

  out_gemm_kernel<<<dim3(NTOK / 64, CCH / 16, BATCH), 128, 0, stream>>>(
      wout_b, O2Tw, b_out, x, (float*)d_out);
}
